// MotionAttentionModel_31662498906982
// MI455X (gfx1250) — compile-verified, matrix operand order checked by reading, not run on hardware
//
#include <hip/hip_runtime.h>
#include <hip/hip_bf16.h>

typedef __attribute__((ext_vector_type(16))) _Float16 v16h;
typedef __attribute__((ext_vector_type(8)))  float    v8f;
typedef __attribute__((ext_vector_type(4)))  float    f4v;
typedef __attribute__((ext_vector_type(2)))  unsigned int u2v;
typedef __attribute__((ext_vector_type(4)))  unsigned int u4v;
typedef __attribute__((ext_vector_type(4)))  int i4v;

union FragU { v16h v; f4v f[2]; _Float16 h[16]; };

#if defined(__has_builtin)
#if __has_builtin(__builtin_amdgcn_global_load_async_to_lds_b128)
#define USE_ASYNC_LDS 1
#else
#define USE_ASYNC_LDS 0
#endif
#else
#define USE_ASYNC_LDS 0
#endif

// ---------------------------------------------------------------------------
// Fragment loaders. gfx1250 16-bit A-matrix (16x32) layout:
// lane l (r=l&15, h=l>>4) holds row r, k in {h*8..h*8+7} U {16+h*8..16+h*8+7}.
// B (32x16) fragments use identical per-lane chunks when the matrix is stored
// n-major ([n][k]) — all weights are pre-transposed that way, and K rows
// (for Q@K^T) are naturally [s][h] = n-major.
// ---------------------------------------------------------------------------
__device__ __forceinline__ v16h frag_lds(const _Float16* base, int row0, int ld, int k0) {
  int l = threadIdx.x & 31, r = l & 15, h = l >> 4;
  const _Float16* p = base + (row0 + r) * ld + k0 + h * 8;
  FragU f;
  f.f[0] = *(const f4v*)(p);
  f.f[1] = *(const f4v*)(p + 16);
  return f.v;
}

__device__ __forceinline__ v16h frag_g(const _Float16* base, long row0, int ld, int k0, long rmax) {
  int l = threadIdx.x & 31, r = l & 15, h = l >> 4;
  long row = row0 + r;
  FragU f;
  if (row < rmax) {
    const _Float16* p = base + row * (long)ld + k0 + h * 8;
    f.f[0] = *(const f4v*)(p);
    f.f[1] = *(const f4v*)(p + 16);
  } else {
    f.f[0] = (f4v)0.f; f.f[1] = (f4v)0.f;
  }
  return f.v;
}

// A-fragment from the raw f32 input x[row][20], zero-padded K 20->32.
__device__ __forceinline__ v16h frag_x(const float* x, long row0, long rmax) {
  int l = threadIdx.x & 31, r = l & 15, h = l >> 4;
  long row = row0 + r;
  FragU f;
#pragma unroll
  for (int i = 0; i < 8; ++i) {
    int k0 = h * 8 + i;        // < 16 -> always valid
    int k1 = 16 + h * 8 + i;   // may exceed 19
    float a = 0.f, b = 0.f;
    if (row < rmax) {
      a = x[row * 20 + k0];
      if (k1 < 20) b = x[row * 20 + k1];
    }
    f.h[i]     = (_Float16)a;
    f.h[8 + i] = (_Float16)b;
  }
  return f.v;
}

__device__ __forceinline__ v8f wmma16(v16h a, v16h b, v8f c) {
  return __builtin_amdgcn_wmma_f32_16x16x32_f16(false, a, false, b, (short)0, c, false, false);
}

// ---------------------------------------------------------------------------
// Prep: transpose+convert W [kin][nout] f32 -> WT [nout][kpad] f16 (zero pad)
// ---------------------------------------------------------------------------
__global__ void wt_kernel(const float* __restrict__ W, _Float16* __restrict__ WT,
                          int kin, int nout, int kpad) {
  int idx = blockIdx.x * 256 + threadIdx.x;
  if (idx >= nout * kpad) return;
  int n = idx / kpad, k = idx - n * kpad;
  WT[idx] = (k < kin) ? (_Float16)W[k * nout + n] : (_Float16)0.f;
}

__global__ void bias_pack(const float* qb1, const float* kb1, const float* vb1,
                          const float* qb2, const float* kb2, const float* vb2,
                          float* b1s, float* b2s) {
  int i = threadIdx.x;
  b1s[i] = qb1[i]; b1s[256 + i] = kb1[i]; b1s[512 + i] = vb1[i];
  b2s[i] = qb2[i]; b2s[256 + i] = kb2[i]; b2s[512 + i] = vb2[i];
}

// Fold IDCT into the last linear: WcT[t][h] = sum_k Wp2[h][k] * M[k][t]
__global__ void wc_kernel(const float* __restrict__ Wp2, const float* __restrict__ pb2,
                          _Float16* __restrict__ WcT, float* __restrict__ bc) {
  int idx = blockIdx.x * 256 + threadIdx.x; // 32*256
  if (idx >= 32 * 256) return;
  int t = idx >> 8, hcol = idx & 255;
  float s = 0.f;
  if (t < 24) {
    for (int k = 0; k < 24; ++k) {
      float ck = (k == 0) ? 0.20412414523193154f : 0.28867513459481287f;
      float Mkt = ck * cosf(3.14159265358979323846f * k * (2.f * t + 1.f) / 48.f);
      s += Wp2[hcol * 24 + k] * Mkt;
    }
  }
  WcT[idx] = (_Float16)s;
  if (hcol == 0 && t < 24) {
    float sb = 0.f;
    for (int k = 0; k < 24; ++k) {
      float ck = (k == 0) ? 0.20412414523193154f : 0.28867513459481287f;
      sb += pb2[k] * ck * cosf(3.14159265358979323846f * k * (2.f * t + 1.f) / 48.f);
    }
    bc[t] = sb;
  }
}

// ---------------------------------------------------------------------------
// QKV: x(K=32 padded) -> relu -> 256, 128 rows/WG, blockIdx.y selects q/k/v
// ---------------------------------------------------------------------------
__global__ void __launch_bounds__(256) qkv_kernel(
    const float* __restrict__ x, const _Float16* __restrict__ W1T_all,
    const _Float16* __restrict__ W2T_all, const float* __restrict__ b1s,
    const float* __restrict__ b2s, _Float16* __restrict__ Qg,
    _Float16* __restrict__ Kg, _Float16* __restrict__ Vg, long Rc) {
  extern __shared__ char smem[];
  _Float16* hs = (_Float16*)smem; // [128][256] f16
  const int m = blockIdx.y;
  const _Float16* W1 = W1T_all + (size_t)m * 8192;   // [256][32]
  const _Float16* W2 = W2T_all + (size_t)m * 65536;  // [256][256]
  const float* b1 = b1s + m * 256;
  const float* b2 = b2s + m * 256;
  _Float16* Out = (m == 0) ? Qg : ((m == 1) ? Kg : Vg);

  const int wave = threadIdx.x >> 5;
  const int l = threadIdx.x & 31, nl = l & 15, hh = l >> 4;
  const long row0 = (long)blockIdx.x * 128 + wave * 16;

  __builtin_prefetch(W2, 0, 1);

  // ---- layer 1: [16x32] x [32x256] ----
  v16h ax = frag_x(x, row0, Rc);
  for (int nt = 0; nt < 16; ++nt) {
    v16h bw = frag_g(W1, nt * 16, 32, 0, 256);
    v8f acc;
#pragma unroll
    for (int r = 0; r < 8; ++r) acc[r] = 0.f;
    acc = wmma16(ax, bw, acc);
    float bias = b1[nt * 16 + nl];
#pragma unroll
    for (int r = 0; r < 8; ++r) {
      float v = fmaxf(acc[r] + bias, 0.f);
      hs[(wave * 16 + r + 8 * hh) * 256 + nt * 16 + nl] = (_Float16)v;
    }
  }
  // wave reads only its own rows of hs: no barrier needed

  // ---- layer 2: [16x256] x [256x256] ----
  v16h a2[8];
#pragma unroll
  for (int kk = 0; kk < 8; ++kk) a2[kk] = frag_lds(hs, wave * 16, 256, kk * 32);
  for (int nt = 0; nt < 16; ++nt) {
    float bias = b2[nt * 16 + nl];
    v8f acc;
#pragma unroll
    for (int r = 0; r < 8; ++r) acc[r] = bias;
#pragma unroll
    for (int kk = 0; kk < 8; ++kk) {
      v16h bw = frag_g(W2, nt * 16, 256, kk * 32, 256);
      acc = wmma16(a2[kk], bw, acc);
    }
#pragma unroll
    for (int r = 0; r < 8; ++r) {
      long row = row0 + r + 8 * hh;
      if (row < Rc) Out[row * 256 + nt * 16 + nl] = (_Float16)acc[r];
    }
  }
}

// ---------------------------------------------------------------------------
// Attention: one WG per batch element, 9 waves x 16 query rows (135 pad 144)
// K tile staged into LDS with GLOBAL_LOAD_ASYNC_TO_LDS (ASYNCcnt) when the
// toolchain exposes the gfx1250 builtins; synchronous fallback otherwise.
// ---------------------------------------------------------------------------
__global__ void __launch_bounds__(288) attn_kernel(
    const _Float16* __restrict__ Qg, const _Float16* __restrict__ Kg,
    const _Float16* __restrict__ Vg, _Float16* __restrict__ Ctx) {
  extern __shared__ char smem[];
  _Float16* Ks  = (_Float16*)smem;            // [144][256]  (s-major: n=s, k=h)
  _Float16* Vt  = Ks + 144 * 256;             // [256][160]  (V transposed: n=h, k=s)
  _Float16* Ps  = Vt + 256 * 160;             // 9 waves * [16][160]
  const int b = blockIdx.x;
  const int tid = threadIdx.x;

#if USE_ASYNC_LDS
  // K rows 0..134: DMA 16B granules straight to LDS, no VGPR round-trip.
  for (int c = tid; c < (135 * 256) / 8; c += 288) {   // 32 granules per row
    int row = c >> 5, col = (c & 31) * 8;
    const _Float16* src = Kg + ((long)b * 135 + row) * 256 + col;
    _Float16* dst = Ks + row * 256 + col;
    __builtin_amdgcn_global_load_async_to_lds_b128(
        (__attribute__((address_space(1))) i4v*)(uintptr_t)src,
        (__attribute__((address_space(3))) i4v*)(unsigned int)(uintptr_t)dst,
        0, 0);
  }
  // zero-fill padded K rows 135..143
  for (int c = tid; c < (9 * 256) / 8; c += 288) {
    u4v z = (u4v)0u;
    *(u4v*)(Ks + 135 * 256 + c * 8) = z;
  }
#else
  for (int c = tid; c < (144 * 256) / 4; c += 288) {
    int row = c >> 6, col = (c * 4) & 255;
    u2v kv = (u2v)0u;
    if (row < 135) kv = *(const u2v*)(Kg + ((long)b * 135 + row) * 256 + col);
    *(u2v*)(Ks + c * 4) = kv;
  }
#endif
  // cooperative transposed load of V: Vt[h][s], zero-padded s in [135,160)
  for (int c = tid; c < 160 * 256; c += 288) {
    int s = c >> 8, hcol = c & 255;
    _Float16 v = (_Float16)0.f;
    if (s < 135) v = Vg[((long)b * 135 + s) * 256 + hcol];
    Vt[hcol * 160 + s] = v;
  }
#if USE_ASYNC_LDS
#if __has_builtin(__builtin_amdgcn_s_wait_asynccnt)
  __builtin_amdgcn_s_wait_asynccnt(0);
#else
  asm volatile("s_wait_asynccnt 0" ::: "memory");
#endif
#endif
  __syncthreads();

  const int wave = tid >> 5;
  const int l = tid & 31, nl = l & 15, hh = l >> 4;
  const long qrow0 = (long)b * 135 + wave * 16;
  const long qmax  = (long)b * 135 + 135;

  // ---- scores: Q[16x256] @ K^T -> 9 tiles of 16x16 ----
  v8f s[9];
#pragma unroll
  for (int st = 0; st < 9; ++st)
#pragma unroll
    for (int r = 0; r < 8; ++r) s[st][r] = 0.f;
  for (int kk = 0; kk < 8; ++kk) {
    v16h aq = frag_g(Qg, qrow0, 256, kk * 32, qmax);
#pragma unroll
    for (int st = 0; st < 9; ++st) {
      v16h bk = frag_lds(Ks, st * 16, 256, kk * 32);
      s[st] = wmma16(aq, bk, s[st]);
    }
  }

  // ---- row softmax over 135 valid columns ----
  const float scale = 0.0625f; // 1/sqrt(256)
  _Float16* Pw = Ps + wave * (16 * 160);
#pragma unroll
  for (int r = 0; r < 8; ++r) {
    float mx = -3.0e38f;
#pragma unroll
    for (int st = 0; st < 9; ++st) {
      if (st * 16 + nl < 135) mx = fmaxf(mx, s[st][r] * scale);
    }
    mx = fmaxf(mx, __shfl_xor(mx, 1, 32));
    mx = fmaxf(mx, __shfl_xor(mx, 2, 32));
    mx = fmaxf(mx, __shfl_xor(mx, 4, 32));
    mx = fmaxf(mx, __shfl_xor(mx, 8, 32));
    float e[9];
    float sum = 0.f;
#pragma unroll
    for (int st = 0; st < 9; ++st) {
      float v = (st * 16 + nl < 135) ? __expf(s[st][r] * scale - mx) : 0.f;
      e[st] = v; sum += v;
    }
    sum += __shfl_xor(sum, 1, 32);
    sum += __shfl_xor(sum, 2, 32);
    sum += __shfl_xor(sum, 4, 32);
    sum += __shfl_xor(sum, 8, 32);
    float inv = 1.f / sum;
    int mrow = r + 8 * hh;
#pragma unroll
    for (int st = 0; st < 9; ++st) Pw[mrow * 160 + st * 16 + nl] = (_Float16)(e[st] * inv);
    Pw[mrow * 160 + 144 + nl] = (_Float16)0.f; // zero pad cols 144..159
  }

  // ---- context: P[16x160] @ V[160x256] ----
  v16h pa[5];
#pragma unroll
  for (int kt = 0; kt < 5; ++kt) pa[kt] = frag_lds(Pw, 0, 160, kt * 32);
  for (int nt = 0; nt < 16; ++nt) {
    v8f acc;
#pragma unroll
    for (int r = 0; r < 8; ++r) acc[r] = 0.f;
#pragma unroll
    for (int kt = 0; kt < 5; ++kt) {
      v16h bv = frag_lds(Vt, nt * 16, 160, kt * 32);
      acc = wmma16(pa[kt], bv, acc);
    }
#pragma unroll
    for (int r = 0; r < 8; ++r) {
      long row = qrow0 + r + 8 * hh;
      if (row < qmax) Ctx[row * 256 + nt * 16 + nl] = (_Float16)acc[r];
    }
  }
}

// ---------------------------------------------------------------------------
// Projection + folded IDCT: relu([Q|ctx]@Wp1+b) @ Wc + bc, scatter to output
// ---------------------------------------------------------------------------
__global__ void __launch_bounds__(256) proj_kernel(
    const _Float16* __restrict__ Qg, const _Float16* __restrict__ Ctx,
    const _Float16* __restrict__ Wp1T, const float* __restrict__ bp1,
    const _Float16* __restrict__ WcT, const float* __restrict__ bc,
    float* __restrict__ out, long Rc, long rowBase) {
  extern __shared__ char smem[];
  _Float16* hs = (_Float16*)smem; // [128][256]
  const int wave = threadIdx.x >> 5;
  const int l = threadIdx.x & 31, nl = l & 15, hh = l >> 4;
  const long row0 = (long)blockIdx.x * 128 + wave * 16;

  __builtin_prefetch(Wp1T, 0, 1);

  v16h aq[8], ac[8];
#pragma unroll
  for (int kk = 0; kk < 8; ++kk) {
    aq[kk] = frag_g(Qg,  row0, 256, kk * 32, Rc);
    ac[kk] = frag_g(Ctx, row0, 256, kk * 32, Rc);
  }
  for (int nt = 0; nt < 16; ++nt) {
    float bias = bp1[nt * 16 + nl];
    v8f acc;
#pragma unroll
    for (int r = 0; r < 8; ++r) acc[r] = bias;
#pragma unroll
    for (int kk = 0; kk < 8; ++kk) {
      v16h bw = frag_g(Wp1T, nt * 16, 512, kk * 32, 256);
      acc = wmma16(aq[kk], bw, acc);
    }
#pragma unroll
    for (int kk = 0; kk < 8; ++kk) {
      v16h bw = frag_g(Wp1T, nt * 16, 512, 256 + kk * 32, 256);
      acc = wmma16(ac[kk], bw, acc);
    }
#pragma unroll
    for (int r = 0; r < 8; ++r)
      hs[(wave * 16 + r + 8 * hh) * 256 + nt * 16 + nl] = (_Float16)fmaxf(acc[r], 0.f);
  }

  v16h ah[8];
#pragma unroll
  for (int kk = 0; kk < 8; ++kk) ah[kk] = frag_lds(hs, wave * 16, 256, kk * 32);
#pragma unroll
  for (int nt = 0; nt < 2; ++nt) {
    int t = nt * 16 + nl;
    float bias = (t < 24) ? bc[t] : 0.f;
    v8f acc;
#pragma unroll
    for (int r = 0; r < 8; ++r) acc[r] = bias;
#pragma unroll
    for (int kk = 0; kk < 8; ++kk) {
      v16h bw = frag_g(WcT, nt * 16, 256, kk * 32, 32);
      acc = wmma16(ah[kk], bw, acc);
    }
    if (t < 24) {
#pragma unroll
      for (int r = 0; r < 8; ++r) {
        long row = row0 + r + 8 * hh;
        if (row < Rc) {
          long grow = rowBase + row;
          long bb = grow / 135;
          long tok = grow - bb * 135;
          out[(bb * 24 + t) * 135 + tok] = acc[r];
        }
      }
    }
  }
}

// ---------------------------------------------------------------------------
extern "C" void kernel_launch(void* const* d_in, const int* in_sizes, int n_in,
                              void* d_out, int out_size, void* d_ws, size_t ws_size,
                              hipStream_t stream) {
  const float* x    = (const float*)d_in[0];
  const float* qW1  = (const float*)d_in[1];  const float* qb1 = (const float*)d_in[2];
  const float* qW2  = (const float*)d_in[3];  const float* qb2 = (const float*)d_in[4];
  const float* kW1  = (const float*)d_in[5];  const float* kb1 = (const float*)d_in[6];
  const float* kW2  = (const float*)d_in[7];  const float* kb2 = (const float*)d_in[8];
  const float* vW1  = (const float*)d_in[9];  const float* vb1 = (const float*)d_in[10];
  const float* vW2  = (const float*)d_in[11]; const float* vb2 = (const float*)d_in[12];
  const float* pW1  = (const float*)d_in[13]; const float* pb1 = (const float*)d_in[14];
  const float* pW2  = (const float*)d_in[15]; const float* pb2 = (const float*)d_in[16];
  float* out = (float*)d_out;

  char* ws = (char*)d_ws;
  _Float16* W1T  = (_Float16*)(ws + 0);        // 3 * [256][32]
  _Float16* W2T  = (_Float16*)(ws + 49152);    // 3 * [256][256]
  _Float16* Wp1T = (_Float16*)(ws + 442368);   // [256][512]
  _Float16* WcT  = (_Float16*)(ws + 704512);   // [32][256]
  float*    b1s  = (float*)(ws + 720896);      // [3][256]
  float*    b2s  = (float*)(ws + 723968);      // [3][256]
  float*    bc   = (float*)(ws + 727040);      // [24]
  const size_t BIG = 1u << 20;
  const long  CHUNK_B = 1024, Rc = CHUNK_B * 135;          // 138240 rows/chunk
  const size_t QKVB = (size_t)Rc * 256 * sizeof(_Float16); // ~70.8 MB
  _Float16* Qg = (_Float16*)(ws + BIG);
  _Float16* Kg = (_Float16*)(ws + BIG + QKVB);
  _Float16* Vg = (_Float16*)(ws + BIG + 2 * QKVB);
  _Float16* Ctx = Kg; // alias: attn reads K into LDS (barrier) before writing ctx

  const int LDS_QKV  = 128 * 256 * 2;                              // 64 KB
  const int LDS_ATTN = (144 * 256 + 256 * 160 + 9 * 16 * 160) * 2; // ~197 KB
  const int LDS_PROJ = 128 * 256 * 2;                              // 64 KB
  (void)hipFuncSetAttribute((const void*)qkv_kernel,  hipFuncAttributeMaxDynamicSharedMemorySize, LDS_QKV);
  (void)hipFuncSetAttribute((const void*)attn_kernel, hipFuncAttributeMaxDynamicSharedMemorySize, LDS_ATTN);
  (void)hipFuncSetAttribute((const void*)proj_kernel, hipFuncAttributeMaxDynamicSharedMemorySize, LDS_PROJ);

  // ---- weight prep ----
  wt_kernel<<<32,  256, 0, stream>>>(qW1, W1T + 0 * 8192,  20, 256, 32);
  wt_kernel<<<32,  256, 0, stream>>>(kW1, W1T + 1 * 8192,  20, 256, 32);
  wt_kernel<<<32,  256, 0, stream>>>(vW1, W1T + 2 * 8192,  20, 256, 32);
  wt_kernel<<<256, 256, 0, stream>>>(qW2, W2T + 0 * 65536, 256, 256, 256);
  wt_kernel<<<256, 256, 0, stream>>>(kW2, W2T + 1 * 65536, 256, 256, 256);
  wt_kernel<<<256, 256, 0, stream>>>(vW2, W2T + 2 * 65536, 256, 256, 256);
  wt_kernel<<<512, 256, 0, stream>>>(pW1, Wp1T, 512, 256, 512);
  wc_kernel<<<32,  256, 0, stream>>>(pW2, pb2, WcT, bc);
  bias_pack<<<1,   256, 0, stream>>>(qb1, kb1, vb1, qb2, kb2, vb2, b1s, b2s);

  // ---- two batch chunks of 1024 ----
  for (int c = 0; c < 2; ++c) {
    const float* xc = x + (size_t)c * CHUNK_B * 135 * 20;
    qkv_kernel<<<dim3(1080, 3), 256, LDS_QKV, stream>>>(xc, W1T, W2T, b1s, b2s,
                                                        Qg, Kg, Vg, Rc);
    attn_kernel<<<dim3((int)CHUNK_B), 288, LDS_ATTN, stream>>>(Qg, Kg, Vg, Ctx);
    proj_kernel<<<dim3(1080), 256, LDS_PROJ, stream>>>(Qg, Ctx, Wp1T, pb1, WcT, bc,
                                                       out, Rc, (long)c * Rc);
  }
}
